// TinyViTBlock_61753039782137
// MI455X (gfx1250) — matrix operand order repaired and checked by reading, not compiled, not run on hardware
//
#include <hip/hip_runtime.h>
#include <hip/hip_bf16.h>
#include <math.h>

typedef __attribute__((ext_vector_type(16))) _Float16 v16h;
typedef __attribute__((ext_vector_type(8)))  _Float16 v8h;
typedef __attribute__((ext_vector_type(8)))  float    v8f;

#define DIM 192
#define NH  6
#define HID 768
#define QKV_OUT 576
#define LDA 200      // f16 row stride for 192-wide LDS tiles (gcd(400,256)=16 -> conflict-free)
#define LDQ 40       // Q stride (80B, 16B-mult for A-frags)
#define LDK 48       // K stride (96B, 32B-mult for B-frags)
#define LDV 80       // V^T stride (160B, 32B-mult for B-frags)
#define LDP 72       // P stride (144B, 16B-mult for A-frags)
#define LDG 792      // fc1 act stride (1584B, 16B-mult, gcd=16 -> conflict-free)

// ---------- WMMA helpers (CDNA5 16x16x32 f16 -> f32) ----------
__device__ __forceinline__ v8f wmma32(v16h a, v16h b, v8f c) {
  return __builtin_amdgcn_wmma_f32_16x16x32_f16(false, a, false, b, (short)0, c, false, false);
}

// A matrix 16x32 f16, row-major [M][lda] source.
// Contiguous-k layout used here (matches bfrag): lane m = lane&15; halfs 0..15 -> K = kbase + 16*hi16 + i
__device__ __forceinline__ v16h afrag(const _Float16* A, int lda, int mt, int kbase, int lane) {
  const _Float16* p = A + (mt * 16 + (lane & 15)) * lda + kbase + 16 * (lane >> 4);
  v8h c0 = *(const v8h*)p;
  v8h c1 = *(const v8h*)(p + 8);
  v16h r;
#pragma unroll
  for (int i = 0; i < 8; ++i) { r[i] = c0[i]; r[i + 8] = c1[i]; }
  return r;
}

// B matrix 32x16 f16 from transposed storage Bt[N][ldk] (K contiguous per column).
// Contiguous-k layout used here (matches afrag): lane n = lane&15; halfs 0..15 -> K = kbase + 16*hi16 + i  (32B contiguous)
__device__ __forceinline__ v16h bfrag(const _Float16* Bt, int ldk, int nt, int kbase, int lane) {
  const _Float16* p = Bt + (nt * 16 + (lane & 15)) * ldk + kbase + 16 * (lane >> 4);
  return *(const v16h*)p;
}

// ---------- prep kernels ----------
__global__ void k_transpose_f16(const float* __restrict__ in, _Float16* __restrict__ out,
                                int K, int N) {
  int i = blockIdx.x * 256 + threadIdx.x;
  if (i >= K * N) return;
  int k = i / N, n = i % N;
  out[n * K + k] = (_Float16)in[k * N + n];
}

// biasPad[6][64][64]: cols>=49 -> -1e30 (mask pad keys); rows>=49 copy row 48 (finite, discarded)
__global__ void k_prep_bias(const float* __restrict__ ab, const int* __restrict__ idxs,
                            float* __restrict__ out, int n_off) {
  int i = blockIdx.x * 256 + threadIdx.x;
  if (i >= 6 * 64 * 64) return;
  int h = i >> 12, r = (i >> 6) & 63, c = i & 63;
  float v;
  if (c < 49) { int rr = r < 49 ? r : 48; v = ab[h * n_off + idxs[rr * 49 + c]]; }
  else        { v = -1e30f; }
  out[i] = v;
}

// ---------- stage 1: fused windowed attention ----------
// LDS byte offsets (Sb aliases XN: XN dead after qkv GEMM, barrier in between)
#define O_XN   0                       // [64][LDA] f16       25600
#define O_S    0                       // [64][64]  f32       16384 (alias)
#define O_Q    25600                   // [6][64][LDQ] f16    30720
#define O_K    56320                   // [6][64][LDK] f16    36864
#define O_VT   93184                   // [6][32][LDV] f16    30720
#define O_P    123904                  // [64][LDP] f16        9216
#define O_AO   133120                  // [64][LDA] f16       25600
#define O_RA   158720                  // [64][4]   f32        1024
#define O_RB   159744                  // [64][4]   f32        1024
#define SMEM_ATTN 160768

__global__ __launch_bounds__(256) void k_attn(
    const float* __restrict__ x, const float* __restrict__ ln1w, const float* __restrict__ ln1b,
    const _Float16* __restrict__ qkvwt, const float* __restrict__ qkvb,
    const float* __restrict__ biasPad,
    const _Float16* __restrict__ projwt, const float* __restrict__ projb,
    float* __restrict__ y1) {
  extern __shared__ char smem[];
  _Float16* XN  = (_Float16*)(smem + O_XN);
  float*    Sb  = (float*)   (smem + O_S);
  _Float16* Qs  = (_Float16*)(smem + O_Q);
  _Float16* Ks  = (_Float16*)(smem + O_K);
  _Float16* VTs = (_Float16*)(smem + O_VT);
  _Float16* Pb  = (_Float16*)(smem + O_P);
  _Float16* AO  = (_Float16*)(smem + O_AO);
  float*    RA  = (float*)   (smem + O_RA);
  float*    RB  = (float*)   (smem + O_RB);

  const int tid = threadIdx.x, lane = tid & 31;
  const int wave = __builtin_amdgcn_readfirstlane(tid >> 5);   // scalarize wave id
  const int hi = lane >> 4;
  const int win = blockIdx.x;
  const int b = win / 100, iy = (win / 10) % 10, ix = win % 10;

  // warm L2 with the weight panels (global_prefetch_b8)
  for (int i = tid; i < (QKV_OUT * DIM * 2 + DIM * DIM * 2) / 256; i += 256)
    __builtin_prefetch((const char*)qkvwt + i * 256, 0, 1);

  // ---- LN1 -> XN (64 padded tokens x 192ch, f16) ----
  {
    int token = tid >> 2, part = tid & 3;     // 4 threads x 48 ch per token
    bool act = token < 49;
    bool valid = false;
    const float* xp = nullptr;
    if (act) {
      int py = iy * 7 + token / 7, px = ix * 7 + token % 7;
      valid = (py < 64) && (px < 64);
      if (valid) xp = x + ((size_t)b * 4096 + py * 64 + px) * DIM;
    }
    float s = 0.f, q = 0.f;
    if (valid) {
      const float4* p4 = (const float4*)(xp + part * 48);
#pragma unroll
      for (int i = 0; i < 12; ++i) {
        float4 v = p4[i];
        s += v.x + v.y + v.z + v.w;
        q += v.x * v.x + v.y * v.y + v.z * v.z + v.w * v.w;
      }
    }
    RA[token * 4 + part] = s; RB[token * 4 + part] = q;
    __syncthreads();
    float ts = RA[token * 4] + RA[token * 4 + 1] + RA[token * 4 + 2] + RA[token * 4 + 3];
    float tq = RB[token * 4] + RB[token * 4 + 1] + RB[token * 4 + 2] + RB[token * 4 + 3];
    float mean = ts * (1.f / 192.f);
    float var  = tq * (1.f / 192.f) - mean * mean;
    float inv  = rsqrtf(var + 1e-5f);
    _Float16* row = XN + token * LDA + part * 48;
    if (!act) {
      for (int i = 0; i < 48; ++i) row[i] = (_Float16)0.f;
    } else if (!valid) {                       // zero-padded pixel: LN(0) = b
      for (int i = 0; i < 48; ++i) row[i] = (_Float16)ln1b[part * 48 + i];
    } else {
      const float4* p4 = (const float4*)(xp + part * 48);
#pragma unroll
      for (int i = 0; i < 12; ++i) {
        float4 v = p4[i];
        int c = part * 48 + i * 4;
        row[i * 4 + 0] = (_Float16)((v.x - mean) * inv * ln1w[c + 0] + ln1b[c + 0]);
        row[i * 4 + 1] = (_Float16)((v.y - mean) * inv * ln1w[c + 1] + ln1b[c + 1]);
        row[i * 4 + 2] = (_Float16)((v.z - mean) * inv * ln1w[c + 2] + ln1b[c + 2]);
        row[i * 4 + 3] = (_Float16)((v.w - mean) * inv * ln1w[c + 3] + ln1b[c + 3]);
      }
    }
  }
  __syncthreads();

  // ---- qkv GEMM: [64x192] @ [192x576], A-frags hoisted (one mt per wave) ----
  const float qscale = 0.17677669529663687f;   // 1/sqrt(32)
  {
    int mt = wave >> 1;
    int nt0 = (wave & 1) * 18;
    v16h a[6];
#pragma unroll
    for (int ks = 0; ks < 6; ++ks) a[ks] = afrag(XN, LDA, mt, ks * 32, lane);
    for (int t = 0; t < 18; ++t) {
      int nt = nt0 + t;
      v8f acc = {};
#pragma unroll
      for (int ks = 0; ks < 6; ++ks)
        acc = wmma32(a[ks], bfrag(qkvwt, DIM, nt, ks * 32, lane), acc);
      int jn = nt * 16, head = jn / 96, off = jn % 96;   // all wave-uniform
      int sect = off >> 5;                               // 0=q 1=k 2=v
      int d = (off & 31) + (lane & 15);
      float bias = qkvb[jn + (lane & 15)];
      if (sect == 0) {
#pragma unroll
        for (int r = 0; r < 8; ++r)
          Qs[(head * 64 + mt * 16 + r + 8 * hi) * LDQ + d] = (_Float16)((acc[r] + bias) * qscale);
      } else if (sect == 1) {
#pragma unroll
        for (int r = 0; r < 8; ++r)
          Ks[(head * 64 + mt * 16 + r + 8 * hi) * LDK + d] = (_Float16)(acc[r] + bias);
      } else {
#pragma unroll
        for (int r = 0; r < 8; ++r)
          VTs[(head * 32 + d) * LDV + mt * 16 + r + 8 * hi] = (_Float16)(acc[r] + bias);
      }
    }
  }
  __syncthreads();   // also fences XN -> Sb alias reuse

  // ---- per-head attention ----
  for (int h = 0; h < 6; ++h) {
    // S = Q @ K^T + bias  (16 tiles, 2 per wave, single K-step of 32)
#pragma unroll
    for (int t = 0; t < 2; ++t) {
      int idx = wave * 2 + t;
      int mt = idx >> 2, nt = idx & 3;
      v8f acc = {};
      acc = wmma32(afrag(Qs + h * 64 * LDQ, LDQ, mt, 0, lane),
                   bfrag(Ks + h * 64 * LDK, LDK, nt, 0, lane), acc);
      int n = nt * 16 + (lane & 15);
#pragma unroll
      for (int r = 0; r < 8; ++r) {
        int m = mt * 16 + r + 8 * hi;
        Sb[m * 64 + n] = acc[r] + biasPad[(h * 64 + m) * 64 + n];
      }
    }
    __syncthreads();
    // softmax rows (4 threads x 16 cols per row)
    {
      int row = tid >> 2, part = tid & 3;
      float* sr = Sb + row * 64 + part * 16;
      float mx = -3.0e38f;
#pragma unroll
      for (int i = 0; i < 16; ++i) mx = fmaxf(mx, sr[i]);
      RA[row * 4 + part] = mx;
      __syncthreads();
      float rm = fmaxf(fmaxf(RA[row * 4], RA[row * 4 + 1]),
                       fmaxf(RA[row * 4 + 2], RA[row * 4 + 3]));
      float sum = 0.f;
#pragma unroll
      for (int i = 0; i < 16; ++i) { float e = __expf(sr[i] - rm); sr[i] = e; sum += e; }
      RB[row * 4 + part] = sum;
      __syncthreads();
      float inv = 1.f / (RB[row * 4] + RB[row * 4 + 1] + RB[row * 4 + 2] + RB[row * 4 + 3]);
      _Float16* pr = Pb + row * LDP + part * 16;
#pragma unroll
      for (int i = 0; i < 16; ++i) pr[i] = (_Float16)(sr[i] * inv);
    }
    __syncthreads();
    // out = P @ V  (8 tiles, 1 per wave, K=64 -> 2 steps) -> AO[:, h*32 + :]
    {
      int mt = wave >> 1, nt = wave & 1;
      v8f acc = {};
#pragma unroll
      for (int ks = 0; ks < 2; ++ks)
        acc = wmma32(afrag(Pb, LDP, mt, ks * 32, lane),
                     bfrag(VTs + h * 32 * LDV, LDV, nt, ks * 32, lane), acc);
      int c = h * 32 + nt * 16 + (lane & 15);
#pragma unroll
      for (int r = 0; r < 8; ++r)
        AO[(mt * 16 + r + 8 * hi) * LDA + c] = (_Float16)acc[r];
    }
    __syncthreads();
  }

  // ---- proj GEMM: [64x192] @ [192x192], A-frags hoisted, residual, window-reverse ----
  {
    int mt = wave >> 1;
    int nt0 = (wave & 1) * 6;
    v16h a[6];
#pragma unroll
    for (int ks = 0; ks < 6; ++ks) a[ks] = afrag(AO, LDA, mt, ks * 32, lane);
    for (int t = 0; t < 6; ++t) {
      int nt = nt0 + t;
      v8f acc = {};
#pragma unroll
      for (int ks = 0; ks < 6; ++ks)
        acc = wmma32(a[ks], bfrag(projwt, DIM, nt, ks * 32, lane), acc);
      int c = nt * 16 + (lane & 15);
      float pb = projb[c];
#pragma unroll
      for (int r = 0; r < 8; ++r) {
        int m = mt * 16 + r + 8 * hi;
        if (m < 49) {
          int py = iy * 7 + m / 7, px = ix * 7 + m % 7;
          if (py < 64 && px < 64) {
            size_t gi = ((size_t)b * 4096 + py * 64 + px) * DIM + c;
            y1[gi] = x[gi] + acc[r] + pb;
          }
        }
      }
    }
  }
}

// ---------- stage 2: depthwise 3x3 conv + folded BN ----------
__global__ __launch_bounds__(192) void k_dwconv_bn(
    const float* __restrict__ y1, const float* __restrict__ convw,
    const float* __restrict__ bng, const float* __restrict__ bnb,
    const float* __restrict__ bnm, const float* __restrict__ bnv,
    float* __restrict__ out) {
  int row = blockIdx.x;                        // B*64 rows
  int b = row >> 6, py = row & 63;
  int c = threadIdx.x;
  float w[9];
#pragma unroll
  for (int t = 0; t < 9; ++t) w[t] = convw[c * 9 + t];
  float scale = bng[c] * rsqrtf(bnv[c] + 1e-5f);
  float bias  = bnb[c] - bnm[c] * scale;
  const float* base = y1 + (size_t)b * 4096 * DIM;
  for (int px = 0; px < 64; ++px) {
    float acc = 0.f;
#pragma unroll
    for (int dy = -1; dy <= 1; ++dy) {
      int yy = py + dy; if (yy < 0 || yy > 63) continue;
#pragma unroll
      for (int dx = -1; dx <= 1; ++dx) {
        int xx = px + dx; if (xx < 0 || xx > 63) continue;
        acc += base[(yy * 64 + xx) * DIM + c] * w[(dy + 1) * 3 + (dx + 1)];
      }
    }
    out[((size_t)b * 4096 + py * 64 + px) * DIM + c] = acc * scale + bias;
  }
}

// ---------- stage 3: LN2 + fc1 + GELU + fc2 + residual (in-place on d_out) ----------
#define O3_HB 0                                 // [64][LDA] f16   25600
#define O3_G  25600                             // [64][LDG] f16  101376
#define O3_RA 126976
#define O3_RB 128000
#define SMEM_MLP 129024

__global__ __launch_bounds__(256) void k_mlp(
    const float* __restrict__ ln2w, const float* __restrict__ ln2b,
    const _Float16* __restrict__ fc1wt, const float* __restrict__ fc1b,
    const _Float16* __restrict__ fc2wt, const float* __restrict__ fc2b,
    float* __restrict__ io) {
  extern __shared__ char smem[];
  _Float16* HB = (_Float16*)(smem + O3_HB);
  _Float16* G  = (_Float16*)(smem + O3_G);
  float* RA = (float*)(smem + O3_RA);
  float* RB = (float*)(smem + O3_RB);
  const int tid = threadIdx.x, lane = tid & 31;
  const int wave = __builtin_amdgcn_readfirstlane(tid >> 5);
  const int hi = lane >> 4;
  const size_t t0 = (size_t)blockIdx.x * 64;

  // warm L2 with both fc weight panels (global_prefetch_b8)
  for (int i = tid; i < (2 * HID * DIM * 2) / 256; i += 256)
    __builtin_prefetch((const char*)fc1wt + i * 256, 0, 1);

  // LN2
  {
    int token = tid >> 2, part = tid & 3;
    const float* xp = io + (t0 + token) * DIM;
    const float4* p4 = (const float4*)(xp + part * 48);
    float s = 0.f, q = 0.f;
#pragma unroll
    for (int i = 0; i < 12; ++i) {
      float4 v = p4[i];
      s += v.x + v.y + v.z + v.w;
      q += v.x * v.x + v.y * v.y + v.z * v.z + v.w * v.w;
    }
    RA[token * 4 + part] = s; RB[token * 4 + part] = q;
    __syncthreads();
    float ts = RA[token * 4] + RA[token * 4 + 1] + RA[token * 4 + 2] + RA[token * 4 + 3];
    float tq = RB[token * 4] + RB[token * 4 + 1] + RB[token * 4 + 2] + RB[token * 4 + 3];
    float mean = ts * (1.f / 192.f);
    float inv  = rsqrtf(tq * (1.f / 192.f) - mean * mean + 1e-5f);
    _Float16* row = HB + token * LDA + part * 48;
#pragma unroll
    for (int i = 0; i < 12; ++i) {
      float4 v = p4[i];
      int c = part * 48 + i * 4;
      row[i * 4 + 0] = (_Float16)((v.x - mean) * inv * ln2w[c + 0] + ln2b[c + 0]);
      row[i * 4 + 1] = (_Float16)((v.y - mean) * inv * ln2w[c + 1] + ln2b[c + 1]);
      row[i * 4 + 2] = (_Float16)((v.z - mean) * inv * ln2w[c + 2] + ln2b[c + 2]);
      row[i * 4 + 3] = (_Float16)((v.w - mean) * inv * ln2w[c + 3] + ln2b[c + 3]);
    }
  }
  __syncthreads();

  // fc1: [64x192] @ [192x768], A-frags hoisted, exact GELU -> G (f16)
  {
    int mt = wave >> 1;
    int nt0 = (wave & 1) * 24;
    v16h a[6];
#pragma unroll
    for (int ks = 0; ks < 6; ++ks) a[ks] = afrag(HB, LDA, mt, ks * 32, lane);
    for (int t = 0; t < 24; ++t) {
      int nt = nt0 + t;
      v8f acc = {};
#pragma unroll
      for (int ks = 0; ks < 6; ++ks)
        acc = wmma32(a[ks], bfrag(fc1wt, DIM, nt, ks * 32, lane), acc);
      int c = nt * 16 + (lane & 15);
      float fb = fc1b[c];
#pragma unroll
      for (int r = 0; r < 8; ++r) {
        float v = acc[r] + fb;
        float g = 0.5f * v * (1.f + erff(v * 0.70710678118f));
        G[(mt * 16 + r + 8 * hi) * LDG + c] = (_Float16)g;
      }
    }
  }
  __syncthreads();

  // fc2: [64x768] @ [768x192], K-outer with 6 accumulator tiles, residual -> d_out
  {
    int mt = wave >> 1;
    int ntb = (wave & 1) * 6;
    v8f acc[6];
#pragma unroll
    for (int j = 0; j < 6; ++j) acc[j] = {};
    for (int ks = 0; ks < 24; ++ks) {
      v16h a = afrag(G, LDG, mt, ks * 32, lane);
#pragma unroll
      for (int j = 0; j < 6; ++j)
        acc[j] = wmma32(a, bfrag(fc2wt, HID, ntb + j, ks * 32, lane), acc[j]);
    }
#pragma unroll
    for (int j = 0; j < 6; ++j) {
      int c = (ntb + j) * 16 + (lane & 15);
      float fb = fc2b[c];
#pragma unroll
      for (int r = 0; r < 8; ++r) {
        size_t gi = (t0 + mt * 16 + r + 8 * hi) * DIM + c;
        io[gi] = io[gi] + acc[j][r] + fb;       // x (=y2) + h; block owns these tokens
      }
    }
  }
}

// ---------- launcher ----------
extern "C" void kernel_launch(void* const* d_in, const int* in_sizes, int n_in,
                              void* d_out, int out_size, void* d_ws, size_t ws_size,
                              hipStream_t stream) {
  (void)n_in; (void)out_size; (void)ws_size;
  const float* x        = (const float*)d_in[0];
  const float* ln1w     = (const float*)d_in[1];
  const float* ln1b     = (const float*)d_in[2];
  const float* qkv_w    = (const float*)d_in[3];
  const float* qkv_b    = (const float*)d_in[4];
  const float* attn_b   = (const float*)d_in[5];
  const float* proj_w   = (const float*)d_in[6];
  const float* proj_b   = (const float*)d_in[7];
  const float* conv_w   = (const float*)d_in[8];
  const float* bn_g     = (const float*)d_in[9];
  const float* bn_b     = (const float*)d_in[10];
  const float* bn_m     = (const float*)d_in[11];
  const float* bn_v     = (const float*)d_in[12];
  const float* ln2w     = (const float*)d_in[13];
  const float* ln2b     = (const float*)d_in[14];
  const float* fc1_w    = (const float*)d_in[15];
  const float* fc1_b    = (const float*)d_in[16];
  const float* fc2_w    = (const float*)d_in[17];
  const float* fc2_b    = (const float*)d_in[18];
  const int*   bias_idx = (const int*)d_in[19];
  int n_off = in_sizes[5] / NH;

  char* ws = (char*)d_ws;
  size_t o = 0;
  _Float16* qkvwt = (_Float16*)(ws + o); o += (size_t)QKV_OUT * DIM * 2;   // 221184
  _Float16* projwt= (_Float16*)(ws + o); o += (size_t)DIM * DIM * 2;       //  73728
  _Float16* fc1wt = (_Float16*)(ws + o); o += (size_t)HID * DIM * 2;       // 294912
  _Float16* fc2wt = (_Float16*)(ws + o); o += (size_t)DIM * HID * 2;       // 294912
  float*    biasPad = (float*)(ws + o);  o += (size_t)NH * 64 * 64 * 4;    //  98304
  float*    y1 = (float*)(ws + o);                                         //  ~50.3 MB

  k_transpose_f16<<<(DIM * QKV_OUT + 255) / 256, 256, 0, stream>>>(qkv_w, qkvwt, DIM, QKV_OUT);
  k_transpose_f16<<<(DIM * DIM + 255) / 256, 256, 0, stream>>>(proj_w, projwt, DIM, DIM);
  k_transpose_f16<<<(DIM * HID + 255) / 256, 256, 0, stream>>>(fc1_w, fc1wt, DIM, HID);
  k_transpose_f16<<<(HID * DIM + 255) / 256, 256, 0, stream>>>(fc2_w, fc2wt, HID, DIM);
  k_prep_bias<<<(NH * 64 * 64 + 255) / 256, 256, 0, stream>>>(attn_b, bias_idx, biasPad, n_off);

  k_attn<<<1600, 256, SMEM_ATTN, stream>>>(x, ln1w, ln1b, qkvwt, qkv_b, biasPad,
                                           projwt, proj_b, y1);
  k_dwconv_bn<<<16 * 64, 192, 0, stream>>>(y1, conv_w, bn_g, bn_b, bn_m, bn_v, (float*)d_out);
  k_mlp<<<1024, 256, SMEM_MLP, stream>>>(ln2w, ln2b, fc1wt, fc1_b, fc2wt, fc2_b, (float*)d_out);
}
